// Net_26319559590355
// MI455X (gfx1250) — matrix operand order repaired and checked by reading, not compiled, not run on hardware
//
#include <hip/hip_runtime.h>
#include <stdint.h>

typedef _Float16 v8h  __attribute__((ext_vector_type(8)));
typedef _Float16 v16h __attribute__((ext_vector_type(16)));
typedef float    v8f  __attribute__((ext_vector_type(8)));
typedef unsigned int u32x4 __attribute__((ext_vector_type(4)));
typedef unsigned int u32x8 __attribute__((ext_vector_type(8)));

#define IMG  8192
#define OUTW 4096

// ---------------- LDS layout (bytes) ----------------
// Region R0 is reused: first as the raw f32 input staging tile (S0, 78x78 f32
// = 24336B, packed by TDM), later as C2 + P + C3 (all fully rewritten before
// being read, so no stale/NaN data can reach a WMMA input).
#define S0_OFF   0
#define C2_OFF   0        // conv2 out, f16  80 x 80   (12800 B)
#define P_OFF    12800    // pool out,  f16  50 x 64   ( 6400 B)
#define C3_OFF   19200    // conv3 out, f16  48 x 48   ( 4608 B)
#define IN_OFF   24336    // input f16      82 x 96    (15744 B)
#define C1_OFF   40080    // conv1 out, f16 82 x 96    (15744 B)
#define W_OFF    55824    // 4 x 9 f32 weights          ( 144 B)
#define BF0_OFF  55968    // B fragments slot 0: [3][32][16] f16 (3072 B)
#define BF1_OFF  59040    // B fragments slot 1: [3][32][16] f16 (3072 B)
#define LDS_BYTES 62112

#define IN_STRIDE 96
#define C1_STRIDE 96
#define C2_STRIDE 80
#define P_STRIDE  64
#define C3_STRIDE 48

// Fill one B-fragment slot for a conv:  B_k[kappa][n] = w[k][kappa-n] for
// kappa-n in {0,1,2}, else 0.  Stored per-lane in the exact register layout
// the WMMA B operand wants: f16 B 32x16 striping -> lanes 0-15 hold K=0..7,16..23,
// lanes 16-31 hold K=8..15,24..31; N = lane%16.  Layout: bf[(k*32 + lane)*16 + e].
// Branch-free: clamped LDS index + one select, so no exec-mask scalarization.
__device__ __forceinline__ void fill_B_lds(_Float16* bf, const float* w9,
                                           int tid) {
  for (int i = tid; i < 3 * 32 * 16; i += 256) {
    int k   = i >> 9;
    int l   = (i >> 4) & 31;
    int e   = i & 15;
    int n   = l & 15;
    int kbB = (l >> 4) << 3;           // 0 or 8
    int d   = ((e >> 3) << 4) + kbB + (e & 7) - n;
    int dc  = d < 0 ? 0 : (d > 2 ? 2 : d);
    float v = w9[k * 3 + dc];
    bf[i] = (_Float16)((d == dc) ? v : 0.0f);
  }
}

// Load the 3 per-lane B fragments for this conv (6 aligned ds_load_b128).
__device__ __forceinline__ void load_B(v16h B[3], const _Float16* bf,
                                       int lane) {
#pragma unroll
  for (int k = 0; k < 3; ++k) {
    const _Float16* bp = bf + (k * 32 + lane) * 16;
    v8h lo = *(const v8h*)(bp);
    v8h hh = *(const v8h*)(bp + 8);
    B[k] = __builtin_shufflevector(lo, hh,
           0,1,2,3,4,5,6,7,8,9,10,11,12,13,14,15);
  }
}

// One 3x3 SAME conv stage over an ntr x ntc grid of 16x16 output tiles.
// Input array origin is (out origin - 1) in both dims, so tile (tr,tc)
// reads input rows tr*16 + i + k, cols tc*16 + kappa.  A f16 16x32 layout:
// lane<16 holds K=0..7,16..23; lane>=16 holds K=8..15,24..31 (two b128s).
__device__ __forceinline__ void conv_f16(
    const _Float16* src, int sstride,
    _Float16* dst, int dstride,
    int ntr, int ntc, const _Float16* bf, int wave, int lane)
{
  const int n   = lane & 15;
  const int hi  = lane >> 4;
  const int kbA = hi << 3;            // 0 or 8
  v16h B[3];
  load_B(B, bf, lane);
  const int nt = ntr * ntc;
  for (int t = wave; t < nt; t += 8) {
    int tr = t / ntc, tc = t - tr * ntc;
    v8f acc = {0.f, 0.f, 0.f, 0.f, 0.f, 0.f, 0.f, 0.f};
#pragma unroll
    for (int k = 0; k < 3; ++k) {
      const _Float16* rp = src + (tr * 16 + n + k) * sstride + tc * 16 + kbA;
      v8h lo = *(const v8h*)(rp);
      v8h hh = *(const v8h*)(rp + 16);
      v16h a = __builtin_shufflevector(lo, hh,
               0,1,2,3,4,5,6,7,8,9,10,11,12,13,14,15);
      acc = __builtin_amdgcn_wmma_f32_16x16x32_f16(
                false, a, false, B[k], (short)0, acc, false, false);
    }
    int rb = tr * 16 + hi * 8;
    int cb = tc * 16 + n;
#pragma unroll
    for (int v = 0; v < 8; ++v)
      dst[(rb + v) * dstride + cb] = (_Float16)acc[v];
  }
}

// Final conv: same math, f32 accumulators go straight to global memory.
__device__ __forceinline__ void conv_f16_out(
    const _Float16* src, int sstride,
    float* __restrict__ out, int oy0, int ox0,
    const _Float16* bf, int wave, int lane)
{
  const int n   = lane & 15;
  const int hi  = lane >> 4;
  const int kbA = hi << 3;
  v16h B[3];
  load_B(B, bf, lane);
  for (int t = wave; t < 4; t += 8) {
    int tr = t >> 1, tc = t & 1;
    v8f acc = {0.f, 0.f, 0.f, 0.f, 0.f, 0.f, 0.f, 0.f};
#pragma unroll
    for (int k = 0; k < 3; ++k) {
      const _Float16* rp = src + (tr * 16 + n + k) * sstride + tc * 16 + kbA;
      v8h lo = *(const v8h*)(rp);
      v8h hh = *(const v8h*)(rp + 16);
      v16h a = __builtin_shufflevector(lo, hh,
               0,1,2,3,4,5,6,7,8,9,10,11,12,13,14,15);
      acc = __builtin_amdgcn_wmma_f32_16x16x32_f16(
                false, a, false, B[k], (short)0, acc, false, false);
    }
    int rb = oy0 + tr * 16 + hi * 8;
    int cb = ox0 + tc * 16 + n;
#pragma unroll
    for (int v = 0; v < 8; ++v)
      out[(size_t)(rb + v) * OUTW + cb] = acc[v];
  }
}

extern "C" __global__ __launch_bounds__(256)
void fused_conv_pool(const float* __restrict__ x,
                     const float* __restrict__ w1,
                     const float* __restrict__ w2,
                     const float* __restrict__ w3,
                     const float* __restrict__ w4,
                     float* __restrict__ out)
{
  extern __shared__ __align__(16) char smem[];
  const int tid  = threadIdx.x;
  const int lane = tid & 31;
  const int wave = __builtin_amdgcn_readfirstlane(tid >> 5);
  const int bx = blockIdx.x & 127;
  const int by = blockIdx.x >> 7;

  float* wlds = (float*)(smem + W_OFF);
  _Float16* bf0 = (_Float16*)(smem + BF0_OFF);
  _Float16* bf1 = (_Float16*)(smem + BF1_OFF);

  // Phase 0: zero working LDS (WMMA overread regions must be finite, never
  // NaN) with b128 stores; stage the 36 weights into LDS.
  {
    u32x4 zz = {0u, 0u, 0u, 0u};
    u32x4* z = (u32x4*)smem;
    for (int i = tid; i < (W_OFF >> 4); i += 256) z[i] = zz;
  }
  if (tid < 36) {
    const float* ws[4] = {w1, w2, w3, w4};
    wlds[tid] = ws[tid / 9][tid % 9];
  }
  __syncthreads();

  // Input halo tile geometry (clamped to the image; zeros == SAME padding).
  const int gy0 = by * 64 - 6, gx0 = bx * 64 - 6;
  const int gy0c = gy0 < 0 ? 0 : gy0;
  const int gx0c = gx0 < 0 ? 0 : gx0;
  int gy1 = gy0 + 78; if (gy1 > IMG) gy1 = IMG;
  int gx1 = gx0 + 78; if (gx1 > IMG) gx1 = IMG;
  const int hcl = gy1 - gy0c, wcl = gx1 - gx0c;
  const int roff = gy0c - gy0, coff = gx0c - gx0;

  // Phase 1: TDM async load of the clamped f32 tile into S0 (wave 0) while
  // everyone else builds the conv1 B fragments (disjoint LDS regions).
  fill_B_lds(bf0, wlds + 0, tid);
  if (wave == 0) {
    uint64_t ga   = (uint64_t)(uintptr_t)(x + (size_t)gy0c * IMG + gx0c);
    uint32_t ldsa = (uint32_t)(uintptr_t)(smem + S0_OFF);
    uint32_t td0 = (uint32_t)(IMG - gx0c);   // remaining row elements
    uint32_t td1 = (uint32_t)(IMG - gy0c);   // remaining rows
    u32x4 g0 = { 1u,                         // count=1, no gather
                 ldsa,
                 (uint32_t)ga,
                 (uint32_t)((ga >> 32) & 0x1FFFFFFull) | (2u << 30) }; // type=2
    u32x8 g1;
    g1[0] = 2u << 16;                                   // data_size = 4 bytes
    g1[1] = (td0 & 0xFFFFu) << 16;                      // tensor_dim0[15:0]
    g1[2] = (td0 >> 16) | ((td1 & 0xFFFFu) << 16);      // dim0 hi / dim1 lo
    g1[3] = ((td1 >> 16) & 0xFFFFu) | ((uint32_t)wcl << 16); // tile_dim0
    g1[4] = (uint32_t)hcl;                              // tile_dim1, tile_dim2=0
    g1[5] = (uint32_t)IMG;                              // tensor_dim0_stride
    g1[6] = 0u;
    g1[7] = 0u;
    u32x4 gz = {0u, 0u, 0u, 0u};
    asm volatile("tensor_load_to_lds %0, %1, %2, %3"
                 :: "s"(g0), "s"(g1), "s"(gz), "s"(gz)
                 : "memory");
    __builtin_amdgcn_s_wait_tensorcnt(0);
  }
  __syncthreads();

  // Phase 2: f32 -> f16 into the padded IN buffer (origin coord -6,-6),
  // division-free indexing; also build conv2's B fragments.
  {
    const float* s0 = (const float*)(smem + S0_OFF);
    _Float16* inb = (_Float16*)(smem + IN_OFF);
    for (int i = tid; i < (hcl << 7); i += 256) {
      int r = i >> 7, c = i & 127;
      if (c < wcl)
        inb[(roff + r) * IN_STRIDE + (coff + c)] = (_Float16)s0[r * wcl + c];
    }
  }
  fill_B_lds(bf1, wlds + 9, tid);
  __syncthreads();

  const _Float16* INB = (const _Float16*)(smem + IN_OFF);
  _Float16* C1 = (_Float16*)(smem + C1_OFF);
  _Float16* C2 = (_Float16*)(smem + C2_OFF);
  _Float16* P  = (_Float16*)(smem + P_OFF);
  _Float16* C3 = (_Float16*)(smem + C3_OFF);

  // Phase 3: conv1 (5x5 tiles, out origin coord -5; valid idx <= 75).
  conv_f16(INB, IN_STRIDE, C1, C1_STRIDE, 5, 5, bf0, wave, lane);
  __syncthreads();

  // Phase 4: conv2 (5x5 tiles, out origin coord -4; valid idx <= 73),
  // overlapped with building conv3's B fragments into the free slot.
  conv_f16(C1, C1_STRIDE, C2, C2_STRIDE, 5, 5, bf1, wave, lane);
  fill_B_lds(bf0, wlds + 18, tid);
  __syncthreads();

  // Phase 5: 2x2 maxpool. P idx p <-> pooled coord p-2, reads C2 rows 2p,2p+1.
  // Every P cell is written (invalid cells -> 0) since P aliases old S0 data.
  // Overlapped with building conv4's B fragments.
  for (int i = tid; i < P_STRIDE * 50; i += 256) {
    int r = i >> 6, c = i & 63;
    float m = 0.0f;
    if (r < 37 && c < 37) {
      float a0 = (float)C2[(2 * r)     * C2_STRIDE + 2 * c];
      float a1 = (float)C2[(2 * r)     * C2_STRIDE + 2 * c + 1];
      float a2 = (float)C2[(2 * r + 1) * C2_STRIDE + 2 * c];
      float a3 = (float)C2[(2 * r + 1) * C2_STRIDE + 2 * c + 1];
      m = fmaxf(fmaxf(a0, a1), fmaxf(a2, a3));
    }
    P[i] = (_Float16)m;
  }
  fill_B_lds(bf1, wlds + 27, tid);
  __syncthreads();

  // Phase 6: conv3 (3x3 tiles, out origin coord -1; valid idx <= 34).
  conv_f16(P, P_STRIDE, C3, C3_STRIDE, 3, 3, bf0, wave, lane);
  __syncthreads();

  // Phase 7: conv4 (2x2 tiles) -> f32 directly to global output.
  conv_f16_out(C3, C3_STRIDE, out, by * 32, bx * 32, bf1, wave, lane);
}

extern "C" void kernel_launch(void* const* d_in, const int* in_sizes, int n_in,
                              void* d_out, int out_size, void* d_ws, size_t ws_size,
                              hipStream_t stream) {
  const float* x  = (const float*)d_in[0];
  const float* w1 = (const float*)d_in[1];
  const float* w2 = (const float*)d_in[2];
  const float* w3 = (const float*)d_in[3];
  const float* w4 = (const float*)d_in[4];
  float* out = (float*)d_out;
  (void)in_sizes; (void)n_in; (void)out_size; (void)d_ws; (void)ws_size;
  dim3 grid(128 * 128), block(256);
  hipLaunchKernelGGL(fused_conv_pool, grid, block, LDS_BYTES, stream,
                     x, w1, w2, w3, w4, out);
}
